// ESN_6279242187037
// MI455X (gfx1250) — matrix operand order repaired and checked by reading, not compiled, not run on hardware
//
#include <hip/hip_runtime.h>
#include <hip/hip_bf16.h>
#include <math.h>

// ---------------------------------------------------------------------------
// ESN persistent kernel for MI455X (gfx1250, wave32, WMMA).
//
// x_{t+1} = 0.5*x_t + 0.5*tanh(u_t + x_t @ W),  x:[64,1024], T=2000.
// Latency-bound: one persistent kernel, one grid barrier per step,
// per-step GEMM via v_wmma_f32_16x16x32_bf16 (f32 accumulate).
// ---------------------------------------------------------------------------

typedef __attribute__((ext_vector_type(16))) __bf16 v16bf;
typedef __attribute__((ext_vector_type(8)))  __bf16 v8bf;
typedef __attribute__((ext_vector_type(8)))  float  v8f;

#define B_    64
#define NIN   16
#define T_    2000
#define N_    1024
#define NWG   64        // one WG per 16 N-columns
#define NTHR  256       // 8 wave32

// ---------------------------------------------------------------------------
// Workspace layout:
//   [0      , 256KB) : xf32 buffer A   (float [64][1024])
//   [256KB  , 512KB) : xf32 buffer B
//   [512KB  , 640KB) : xbf16 buffer A  (__bf16 [64][1024])
//   [640KB  , 768KB) : xbf16 buffer B
//   [768KB  , ...  ) : barrier {count, generation}
// ---------------------------------------------------------------------------

__global__ void esn_init(float* xf0, float* xf1, __bf16* xb0, __bf16* xb1,
                         unsigned* bar) {
    int i = blockIdx.x * blockDim.x + threadIdx.x;
    for (int idx = i; idx < B_ * N_; idx += 64 * 256) {
        xf0[idx] = 0.0f; xf1[idx] = 0.0f;
        xb0[idx] = (__bf16)0.0f; xb1[idx] = (__bf16)0.0f;
    }
    if (i == 0) { bar[0] = 0u; bar[1] = 0u; }
}

// Grid-wide arrive/wait barrier through L2 atomics (generation counter).
__device__ __forceinline__ void grid_barrier(unsigned* bar) {
    __threadfence();            // publish this WG's state writes device-wide
    __syncthreads();
    if (threadIdx.x == 0) {
        unsigned gen = __atomic_load_n(bar + 1, __ATOMIC_ACQUIRE);
        if (atomicAdd(bar, 1u) == (unsigned)(NWG - 1)) {
            __atomic_store_n(bar, 0u, __ATOMIC_RELAXED);
            __atomic_store_n(bar + 1, gen + 1u, __ATOMIC_RELEASE);
        } else {
            while (__atomic_load_n(bar + 1, __ATOMIC_ACQUIRE) == gen)
                __builtin_amdgcn_s_sleep(2);
        }
    }
    __syncthreads();
}

__global__ __launch_bounds__(NTHR, 1)
void esn_persistent(const float* __restrict__ Input,   // [64][16][2000]
                    const float* __restrict__ Win,     // [16][1024]
                    const float* __restrict__ W,       // [1024][1024]
                    float* __restrict__ Out,           // [64][1024][2000]
                    float* xf0, float* xf1,
                    __bf16* xb0, __bf16* xb1,
                    unsigned* bar) {
    // B-fragment-major W slice: ldsW[(kblk*16 + n)*16 + j] = W[kblk*16+j][n0+n]
    // (kblk = 16 consecutive K values; 32 contiguous bytes per (kblk, col))
    __shared__ __bf16 ldsW[64 * 16 * 16];      // 32 KB
    __shared__ v8f    lds_red[4][32];          //  4 KB  K-split partial sums
    __shared__ float  lds_in[4][16][16];       //  4 KB  Input[b_tile][i] stage

    const int tid  = threadIdx.x;
    const int wave = tid >> 5;                 // 0..7
    const int lane = tid & 31;
    const int nl   = lane & 15;                // column within 16-wide N tile
    const int hi   = lane >> 4;                // lane half (K/M sub-split)
    const int m    = wave & 3;                 // M tile (16 batches each)
    const int kh   = wave >> 2;                // K half (0: K<512, 1: K>=512)
    const int n0   = blockIdx.x * 16;          // first owned column

    // One-time: convert this WG's W slice f32 -> bf16 into LDS.
    for (int idx = tid; idx < 64 * 16 * 16; idx += NTHR) {
        int kblk = idx >> 8;
        int rem  = idx & 255;
        int n    = rem >> 4;
        int j    = rem & 15;
        ldsW[idx] = (__bf16)W[(size_t)(kblk * 16 + j) * N_ + (n0 + n)];
    }
    // Per-lane input-projection column (constant over t).
    float wcol[NIN];
#pragma unroll
    for (int i = 0; i < NIN; ++i) wcol[i] = Win[i * N_ + n0 + nl];
    __syncthreads();

    const int kbase  = kh * 512;
    const int kblk0  = (kbase >> 4) + hi;      // LDS k-block base for B frags

    for (int t = 0; t < T_; ++t) {
        const int p = t & 1;
        const float*  xcur = p ? xf1 : xf0;
        float*        xnxt = p ? xf0 : xf1;
        const __bf16* xbc  = p ? xb1 : xb0;
        __bf16*       xbn  = p ? xb0 : xb1;

        // ---- GEMM: acc = x[16m..16m+15, kbase:kbase+512] @ W[.., n0:n0+16]
        v8f acc = {};
        // A fragment base: row r = 16m+nl, lane K-offset 16*hi (contiguous k = 16*hi + s).
        const __bf16* xrow =
            xbc + (size_t)(16 * m + nl) * N_ + kbase + hi * 16;
#pragma unroll
        for (int kw = 0; kw < 16; ++kw) {
            union { v16bf v; v8bf h[2]; } A, Bf;
            A.h[0] = *(const v8bf*)(xrow + kw * 32);
            A.h[1] = *(const v8bf*)(xrow + kw * 32 + 8);
            const __bf16* bp = &ldsW[((kblk0 + kw * 2) * 16 + nl) * 16];
            Bf.h[0] = *(const v8bf*)(bp);
            Bf.h[1] = *(const v8bf*)(bp + 8);
            acc = __builtin_amdgcn_wmma_f32_16x16x32_bf16(
                false, A.v, false, Bf.v, (short)0, acc, false, false);
        }

        // ---- K-split reduce + input staging
        if (wave >= 4) {
            lds_red[m][lane] = acc;            // partial for K-high half
#pragma unroll
            for (int q = 0; q < 8; ++q) {      // stage Input[16m..+15][i][t]
                int idx = lane * 8 + q;        // 0..255
                int bl = idx >> 4, i = idx & 15;
                lds_in[m][bl][i] =
                    Input[((size_t)(16 * m + bl) * NIN + i) * T_ + t];
            }
        }
        __syncthreads();

        if (wave < 4) {
            v8f oth = lds_red[m][lane];
#pragma unroll
            for (int v = 0; v < 8; ++v) {
                int   bl = hi * 8 + v;                 // C/D layout: M=v+8*hi
                float s  = acc[v] + oth[v];
                float u  = 0.0f;
#pragma unroll
                for (int i = 0; i < NIN; ++i)
                    u = fmaf(lds_in[m][bl][i], wcol[i], u);
                int   b   = 16 * m + bl;
                int   col = n0 + nl;
                float xp  = xcur[b * N_ + col];
                float xn  = 0.5f * xp + 0.5f * tanhf(s + u);
                xnxt[b * N_ + col] = xn;
                xbn[b * N_ + col]  = (__bf16)xn;
                Out[(size_t)(b * N_ + col) * T_ + t] = xn;   // L2 absorbs
            }
        }
        grid_barrier(bar);
    }
}

extern "C" void kernel_launch(void* const* d_in, const int* in_sizes, int n_in,
                              void* d_out, int out_size, void* d_ws, size_t ws_size,
                              hipStream_t stream) {
    const float* Input = (const float*)d_in[0];   // [64,16,2000]
    const float* Win   = (const float*)d_in[1];   // [16,1024]
    const float* W     = (const float*)d_in[2];   // [1024,1024]
    float*       Out   = (float*)d_out;           // [64,1024,2000]

    char* ws = (char*)d_ws;
    float*    xf0 = (float*)ws;
    float*    xf1 = xf0 + B_ * N_;
    __bf16*   xb0 = (__bf16*)(ws + 512 * 1024);
    __bf16*   xb1 = xb0 + B_ * N_;
    unsigned* bar = (unsigned*)(ws + 768 * 1024);

    esn_init<<<64, 256, 0, stream>>>(xf0, xf1, xb0, xb1, bar);
    esn_persistent<<<NWG, NTHR, 0, stream>>>(Input, Win, W, Out,
                                             xf0, xf1, xb0, xb1, bar);
}
